// ManifoldReconstructionLayer_83107617178202
// MI455X (gfx1250) — matrix operand order repaired and checked by reading, not compiled, not run on hardware
//
#include <hip/hip_runtime.h>

// ---------------------------------------------------------------------------
// ManifoldReconstructionLayer for MI455X (gfx1250, wave32, WMMA f16->f32)
// n = 4096 points, p = 64 features, fp32 in/out.
// ---------------------------------------------------------------------------

typedef __attribute__((ext_vector_type(16))) _Float16 v16h;
typedef __attribute__((ext_vector_type(8)))  _Float16 v8h;
typedef __attribute__((ext_vector_type(8)))  float    v8f;

#define P 64  // feature dimension (from reference setup_inputs)

// ---- WMMA helpers ---------------------------------------------------------

__device__ __forceinline__ v8f wmma_f16(v16h a, v16h b, v8f c) {
  // D = A(16x32 f16) * B(32x16 f16) + C(16x16 f32)
  return __builtin_amdgcn_wmma_f32_16x16x32_f16(
      /*neg_a=*/false, a, /*neg_b=*/false, b,
      /*c_mod=*/(short)0, c, /*reuse_a=*/false, /*reuse_b=*/false);
}

// A operand (16x32 f16): lane l (group g = l>>4, lr = l&15) holds row M=lr,
// elements 0..7 = K[k0 + g*8 .. +7], elements 8..15 = K[k0 + 16 + g*8 .. +7].
__device__ __forceinline__ v16h load_opA(const _Float16* row, int k0, int g) {
  v8h lo = *(const v8h*)(row + k0 + g * 8);
  v8h hi = *(const v8h*)(row + k0 + 16 + g * 8);
  return __builtin_shufflevector(lo, hi, 0, 1, 2, 3, 4, 5, 6, 7,
                                 8, 9, 10, 11, 12, 13, 14, 15);
}

// B operand (32x16 f16) read from a "B^T" buffer stored row-major over K:
// lane holds column N=lr, elements 0..7 = K[k0 + g*8 .. +7], 8..15 = K[k0 + 16 + g*8 .. +7].
__device__ __forceinline__ v16h load_opB(const _Float16* rowT, int k0, int g) {
  v8h lo = *(const v8h*)(rowT + k0 + g * 8);
  v8h hi = *(const v8h*)(rowT + k0 + 16 + g * 8);
  return __builtin_shufflevector(lo, hi, 0, 1, 2, 3, 4, 5, 6, 7,
                                 8, 9, 10, 11, 12, 13, 14, 15);
}

__device__ __forceinline__ float w2f(float d, float r) {
  // reference _weight2
  float half = 0.5f * r;
  if (d < half) return 1.0f;
  if (d < r) {
    float t = (2.0f * d - r) / r;
    float b = 1.0f - t * t;
    return b * b * b;
  }
  return 0.0f;
}

// ---------------------------------------------------------------------------
// K0: sq[i] = ||x_i||^2 ; xh = f16(x) row-major ; xTh = f16(x^T) ; zero maxbuf
// ---------------------------------------------------------------------------
__global__ void k_prep(const float* __restrict__ x, _Float16* __restrict__ xh,
                       _Float16* __restrict__ xTh, float* __restrict__ sq,
                       unsigned int* __restrict__ maxbuf, int n) {
  int i = blockIdx.x * blockDim.x + threadIdx.x;
  if (i == 0) *maxbuf = 0u;
  if (i >= n) return;
  const float* xr = x + (size_t)i * P;
  float s = 0.0f;
  for (int c = 0; c < P; ++c) {
    float v = xr[c];
    s += v * v;
    _Float16 h = (_Float16)v;
    xh[(size_t)i * P + c] = h;
    xTh[(size_t)c * n + i] = h;
  }
  sq[i] = s;
}

// ---------------------------------------------------------------------------
// K1: d2[i,j] = max(sq_i + sq_j - 2*(x_i . x_j), 0)  via WMMA; atomic max.
// Block = 128 threads (4 waves), 64x64 output tile.
// ---------------------------------------------------------------------------
__global__ void __launch_bounds__(128) k_gram(const _Float16* __restrict__ xh,
                                              const float* __restrict__ sq,
                                              float* __restrict__ d2,
                                              unsigned int* __restrict__ maxbuf,
                                              int n) {
  __shared__ __align__(32) _Float16 As[64 * P];
  __shared__ __align__(32) _Float16 Bs[64 * P];
  __shared__ float wmax[4];
  const int t = threadIdx.x, lane = t & 31, wave = t >> 5;
  const int g = lane >> 4, lr = lane & 15;
  const int ib = blockIdx.x * 64, jb = blockIdx.y * 64;

  {  // stage both 64x64 f16 row blocks (contiguous 8 KB each)
    const uint4* sA = (const uint4*)(xh + (size_t)ib * P);
    const uint4* sB = (const uint4*)(xh + (size_t)jb * P);
    uint4* dA = (uint4*)As;
    uint4* dB = (uint4*)Bs;
    for (int k = t; k < 64 * P / 8; k += 128) { dA[k] = sA[k]; dB[k] = sB[k]; }
  }
  __syncthreads();

  const _Float16* arow = As + (wave * 16 + lr) * P;
  v8f zero = {};
  v8f acc[4] = {zero, zero, zero, zero};
  for (int k0 = 0; k0 < P; k0 += 32) {
    v16h a = load_opA(arow, k0, g);
    for (int nt = 0; nt < 4; ++nt) {
      v16h b = load_opB(Bs + (nt * 16 + lr) * P, k0, g);
      acc[nt] = wmma_f16(a, b, acc[nt]);
    }
  }

  float sqi[8];
  for (int r = 0; r < 8; ++r) sqi[r] = sq[ib + wave * 16 + r + g * 8];
  float lmax = 0.0f;
  for (int nt = 0; nt < 4; ++nt) {
    int jcol = jb + nt * 16 + lr;
    float sqj = sq[jcol];
    for (int r = 0; r < 8; ++r) {
      int irow = ib + wave * 16 + r + g * 8;
      float v = fmaxf(sqi[r] + sqj - 2.0f * acc[nt][r], 0.0f);
      d2[(size_t)irow * n + jcol] = v;
      lmax = fmaxf(lmax, v);
    }
  }
  for (int m = 16; m > 0; m >>= 1) lmax = fmaxf(lmax, __shfl_xor(lmax, m, 32));
  if (lane == 0) wmax[wave] = lmax;
  __syncthreads();
  if (t == 0) {
    float m = fmaxf(fmaxf(wmax[0], wmax[1]), fmaxf(wmax[2], wmax[3]));
    atomicMax(maxbuf, __float_as_uint(m));  // valid ordering: all values >= 0
  }
}

// ---------------------------------------------------------------------------
// K2: r clamps -> params = { r0m^2, r1m, r2m }
// ---------------------------------------------------------------------------
__global__ void k_params(const unsigned int* __restrict__ maxbuf,
                         const float* __restrict__ r0p,
                         const float* __restrict__ r1p,
                         const float* __restrict__ r2p,
                         float* __restrict__ params) {
  float md = sqrtf(__uint_as_float(*maxbuf));
  float r0 = fminf(*r0p, md), r1 = fminf(*r1p, md), r2 = fminf(*r2p, md);
  params[0] = r0 * r0;
  params[1] = r1;
  params[2] = r2;
}

// ---------------------------------------------------------------------------
// K3: mu[j] = (sum_i w1(d2[j,i]) x_i) / S1[j] ; U = x - mu ; emit U, Uh.
// w1 weights generated on the fly into LDS A-tile (never hit global memory).
// Block: 64 output rows, full p=64 columns, K-loop over i in chunks of 32.
// ---------------------------------------------------------------------------
__global__ void __launch_bounds__(128) k_mu(const float* __restrict__ x,
                                            const float* __restrict__ d2,
                                            const _Float16* __restrict__ xTh,
                                            const float* __restrict__ params,
                                            float* __restrict__ U,
                                            _Float16* __restrict__ Uh, int n) {
  __shared__ __align__(32) _Float16 Aw[64 * 32];
  __shared__ __align__(32) _Float16 Bt[64 * 32];
  __shared__ float s1arr[128];
  __shared__ float s1row[64];
  const int t = threadIdx.x, lane = t & 31, wave = t >> 5;
  const int g = lane >> 4, lr = lane & 15;
  const int jb = blockIdx.x * 64;
  const float r0m2 = params[0];
  const float inv_r0m2 = (r0m2 > 0.0f) ? 1.0f / r0m2 : 0.0f;

  const int rload = t >> 1;        // 0..63
  const int hload = (t & 1) * 16;  // 0 or 16
  float s1part = 0.0f;

  v8f zero = {};
  v8f acc[4] = {zero, zero, zero, zero};

  for (int i0 = 0; i0 < n; i0 += 32) {
    {  // A-tile: w1(d2[jb+row, i0..i0+31]) -> f16, accumulate row-sum partials
      const float* dr = d2 + (size_t)(jb + rload) * n + i0 + hload;
      _Float16* ar = Aw + rload * 32 + hload;
#pragma unroll
      for (int c = 0; c < 16; ++c) {
        float dv = dr[c];
        float w = 0.0f;
        if (dv < r0m2) {
          float b = 1.0f - dv * inv_r0m2;
          w = b * b * b;
        }
        s1part += w;
        ar[c] = (_Float16)w;
      }
    }
    {  // B^T tile: xTh[f, i0..i0+31], f = 0..63
      const uint4* s = (const uint4*)(xTh + (size_t)rload * n + i0 + hload);
      uint4* d = (uint4*)(Bt + rload * 32 + hload);
      d[0] = s[0];
      d[1] = s[1];
    }
    __syncthreads();
    const _Float16* arow = Aw + (wave * 16 + lr) * 32;
    v16h a = load_opA(arow, 0, g);
    for (int nt = 0; nt < 4; ++nt) {
      v16h b = load_opB(Bt + (nt * 16 + lr) * 32, 0, g);
      acc[nt] = wmma_f16(a, b, acc[nt]);
    }
    __syncthreads();
  }

  s1arr[t] = s1part;
  __syncthreads();
  if (t < 64) s1row[t] = s1arr[2 * t] + s1arr[2 * t + 1];
  __syncthreads();

  for (int nt = 0; nt < 4; ++nt) {
    int col = nt * 16 + lr;
    for (int r = 0; r < 8; ++r) {
      int jr = wave * 16 + r + g * 8;
      int j = jb + jr;
      float mu = acc[nt][r] / s1row[jr];
      float xv = x[(size_t)j * P + col];
      if (mu != mu) mu = xv;  // NaN fallback (reference: where(isnan(mu), y, mu))
      float u = xv - mu;
      U[(size_t)j * P + col] = u;
      Uh[(size_t)j * P + col] = (_Float16)u;
    }
  }
}

// ---------------------------------------------------------------------------
// K4: Udot[j] = ||U_j||^2 ; yU[j] = y_j . U_j
// ---------------------------------------------------------------------------
__global__ void k_udots(const float* __restrict__ x, const float* __restrict__ U,
                        float* __restrict__ Udot, float* __restrict__ yU, int n) {
  int j = blockIdx.x * blockDim.x + threadIdx.x;
  if (j >= n) return;
  float ud = 0.0f, yu = 0.0f;
  for (int c = 0; c < P; ++c) {
    float u = U[(size_t)j * P + c];
    ud += u * u;
    yu += x[(size_t)j * P + c] * u;
  }
  Udot[j] = ud;
  yU[j] = yu;
}

// ---------------------------------------------------------------------------
// K5: fully fused beta + e_Z. Inner WMMA GEMM builds the c-tile
// (c[j,i] = U_j.x_i - yU_j, K=64 features), elementwise builds the beta-weight
// tile in LDS (f16, never written to global), outer WMMA GEMM accumulates
// e_Z_raw over K=n, with row-sums S2 reduced via wave32 shuffles.
// ---------------------------------------------------------------------------
__global__ void __launch_bounds__(128)
k_ez(const float* __restrict__ x, const _Float16* __restrict__ xh,
     const _Float16* __restrict__ xTh, const _Float16* __restrict__ Uh,
     const float* __restrict__ d2, const float* __restrict__ Udot,
     const float* __restrict__ yU, const float* __restrict__ params,
     float* __restrict__ out, int n) {
  __shared__ __align__(32) _Float16 UhS[64 * P];   // U rows (f16), whole block
  __shared__ __align__(32) _Float16 XhC[32 * P];   // x rows for the i-chunk
  __shared__ __align__(32) _Float16 BtC[64 * 32];  // x^T chunk (outer B^T)
  __shared__ __align__(32) _Float16 Wt[64 * 32];   // beta-weight tile (outer A)
  __shared__ float yUs[64], Uds[64], s2row[64];
  const int t = threadIdx.x, lane = t & 31, wave = t >> 5;
  const int g = lane >> 4, lr = lane & 15;
  const int jb = blockIdx.x * 64;
  const float r1m = params[1], r2m = params[2];

  {  // stage U rows (contiguous 8 KB) + per-row scalars
    const uint4* s = (const uint4*)(Uh + (size_t)jb * P);
    uint4* d = (uint4*)UhS;
    for (int k = t; k < 64 * P / 8; k += 128) d[k] = s[k];
  }
  if (t < 64) {
    yUs[t] = yU[jb + t];
    Uds[t] = Udot[jb + t];
  }
  __syncthreads();

  // hoist inner-GEMM A operands (constant for this block)
  const _Float16* urow = UhS + (wave * 16 + lr) * P;
  v16h ua0 = load_opA(urow, 0, g);
  v16h ua1 = load_opA(urow, 32, g);

  v8f zero = {};
  v8f acc[4] = {zero, zero, zero, zero};
  float s2part[8] = {0.f, 0.f, 0.f, 0.f, 0.f, 0.f, 0.f, 0.f};
  const int rload = t >> 1, hload = (t & 1) * 16;

  for (int i0 = 0; i0 < n; i0 += 32) {
    {  // x rows i0..i0+31 (contiguous 4 KB)
      const uint4* s = (const uint4*)(xh + (size_t)i0 * P);
      uint4* d = (uint4*)XhC;
      for (int k = t; k < 32 * P / 8; k += 128) d[k] = s[k];
    }
    {  // x^T chunk rows f=0..63, cols i0..i0+31
      const uint4* s = (const uint4*)(xTh + (size_t)rload * n + i0 + hload);
      uint4* d = (uint4*)(BtC + rload * 32 + hload);
      d[0] = s[0];
      d[1] = s[1];
    }
    __syncthreads();

    // inner GEMM: c[j, i] over 16 rows x 32 cols per wave, K = 64 features
    v8f cacc[2] = {zero, zero};
    for (int nt = 0; nt < 2; ++nt) {
      v16h b0 = load_opB(XhC + (nt * 16 + lr) * P, 0, g);
      cacc[nt] = wmma_f16(ua0, b0, cacc[nt]);
      v16h b1 = load_opB(XhC + (nt * 16 + lr) * P, 32, g);
      cacc[nt] = wmma_f16(ua1, b1, cacc[nt]);
    }

    // elementwise: c -> du/dv -> w2*w2, write beta-weight tile, track row sums
    for (int nt = 0; nt < 2; ++nt) {
      int icol = i0 + nt * 16 + lr;
      for (int r = 0; r < 8; ++r) {
        int jr = wave * 16 + r + g * 8;
        float c = cacc[nt][r] - yUs[jr];
        float du2 = fmaxf(c * c * Uds[jr], 1e-6f);
        float d2v = d2[(size_t)(jb + jr) * n + icol];
        float dvv = sqrtf(fmaxf(d2v - du2, 1e-6f));
        float duu = sqrtf(du2);
        float w = w2f(dvv, r1m) * w2f(duu, r2m);
        s2part[r] += w;
        Wt[jr * 32 + nt * 16 + lr] = (_Float16)w;
      }
    }
    __syncthreads();

    // outer GEMM accumulate: A = beta-weight tile rows (K=32), B^T = x^T chunk
    {
      const _Float16* arow = Wt + (wave * 16 + lr) * 32;
      v16h a = load_opA(arow, 0, g);
      for (int nt = 0; nt < 4; ++nt) {
        v16h b = load_opB(BtC + (nt * 16 + lr) * 32, 0, g);
        acc[nt] = wmma_f16(a, b, acc[nt]);
      }
    }
    __syncthreads();
  }

  // S2 row sums: butterfly within each 16-lane group (matches C-layout rows)
  for (int r = 0; r < 8; ++r) {
    float v = s2part[r];
    for (int m = 1; m < 16; m <<= 1) v += __shfl_xor(v, m, 32);
    if (lr == 0) s2row[wave * 16 + r + g * 8] = v;
  }
  __syncthreads();

  for (int nt = 0; nt < 4; ++nt) {
    int col = nt * 16 + lr;
    for (int r = 0; r < 8; ++r) {
      int jr = wave * 16 + r + g * 8;
      int j = jb + jr;
      float e = acc[nt][r] / s2row[jr];
      float xv = x[(size_t)j * P + col];
      if (e != e) e = xv;  // where(isnan(e_Z), y, e_Z)
      out[(size_t)j * P + col] = e;
    }
  }
}

// ---------------------------------------------------------------------------
// launcher
// ---------------------------------------------------------------------------
extern "C" void kernel_launch(void* const* d_in, const int* in_sizes, int n_in,
                              void* d_out, int out_size, void* d_ws,
                              size_t ws_size, hipStream_t stream) {
  const float* x = (const float*)d_in[0];
  const float* r0 = (const float*)d_in[1];
  const float* r1 = (const float*)d_in[2];
  const float* r2 = (const float*)d_in[3];
  const int n = in_sizes[0] / P;  // 4096

  char* ws = (char*)d_ws;
  size_t off = 0;
  auto alloc = [&](size_t bytes) -> void* {
    void* p = ws + off;
    off = (off + bytes + 255) & ~(size_t)255;
    return p;
  };
  float*        d2v    = (float*)alloc((size_t)n * n * sizeof(float));
  _Float16*     xh     = (_Float16*)alloc((size_t)n * P * 2);
  _Float16*     xTh    = (_Float16*)alloc((size_t)n * P * 2);
  _Float16*     Uhb    = (_Float16*)alloc((size_t)n * P * 2);
  float*        Ub     = (float*)alloc((size_t)n * P * sizeof(float));
  float*        sq     = (float*)alloc((size_t)n * sizeof(float));
  float*        Udot   = (float*)alloc((size_t)n * sizeof(float));
  float*        yU     = (float*)alloc((size_t)n * sizeof(float));
  unsigned int* maxbuf = (unsigned int*)alloc(256);
  float*        params = (float*)alloc(256);

  k_prep<<<(n + 255) / 256, 256, 0, stream>>>(x, xh, xTh, sq, maxbuf, n);
  dim3 gg(n / 64, n / 64);
  k_gram<<<gg, 128, 0, stream>>>(xh, sq, d2v, maxbuf, n);
  k_params<<<1, 1, 0, stream>>>(maxbuf, r0, r1, r2, params);
  k_mu<<<n / 64, 128, 0, stream>>>(x, d2v, xTh, params, Ub, Uhb, n);
  k_udots<<<(n + 255) / 256, 256, 0, stream>>>(x, Ub, Udot, yU, n);
  k_ez<<<n / 64, 128, 0, stream>>>(x, xh, xTh, Uhb, d2v, Udot, yU, params,
                                   (float*)d_out, n);
}
